// Block_60017872995079
// MI455X (gfx1250) — matrix operand order repaired and checked by reading, not compiled, not run on hardware
//
#include <hip/hip_runtime.h>
#include <hip/hip_bf16.h>

// ---------------------------------------------------------------------------
// Transformer block (LN+conv+QKV+RoPE+causal attn+proj, LN+router+SwiGLU MoE)
// for MI455X (gfx1250). Matmuls via v_wmma_f32_16x16x32_bf16 (bf16 in, f32
// acc). A-operand tiles streamed into LDS by the Tensor Data Mover
// (tensor_load_to_lds + s_wait_tensorcnt), double-buffered against compute.
// MoE uses top-1 token gathering (== reference's masked sum).
// ---------------------------------------------------------------------------

typedef __attribute__((ext_vector_type(16))) __bf16 v16bf;
typedef __attribute__((ext_vector_type(8)))  float  v8f;
typedef __attribute__((ext_vector_type(4)))  unsigned int v4u;
typedef __attribute__((ext_vector_type(8)))  int v8i;
typedef __attribute__((ext_vector_type(4)))  int v4i;

#define CB   1024   // channels
#define TT   1024   // sequence
#define BB   4      // batch
#define HH   16     // heads
#define DD   64     // head dim
#define EE   4      // experts
#define DFF  4096
#define MTOK (BB*TT)          // 4096 tokens
#define PADM (MTOK + 512)     // expert segments padded to 128 -> <= 4608 rows
#define MT_TILES (PADM/128)   // 36 M-tiles

__device__ inline v8f zero8() {
  v8f z;
#pragma unroll
  for (int i = 0; i < 8; ++i) z[i] = 0.0f;
  return z;
}

// two contiguous 16-byte LDS loads -> one 16-element bf16 fragment
__device__ inline v16bf ld2x16(const __bf16* p0, const __bf16* p1) {
  v16bf r;
  reinterpret_cast<float4*>(&r)[0] = *reinterpret_cast<const float4*>(p0);
  reinterpret_cast<float4*>(&r)[1] = *reinterpret_cast<const float4*>(p1);
  return r;
}

__device__ inline v8f wmma_bf16(v16bf a, v16bf b, v8f c) {
  return __builtin_amdgcn_wmma_f32_16x16x32_bf16(false, a, false, b, (short)0, c,
                                                 false, false);
}

__device__ inline float warpSum(float v) {
#pragma unroll
  for (int m = 16; m >= 1; m >>= 1) v += __shfl_xor(v, m);
  return v;
}

// ---------------------------------------------------------------------------
// Tensor Data Mover: 2D bf16 tile (tile_d1 rows x tile_d0 elems) from global
// (row stride = stride0 elems) into LDS at lds_addr. D# per ISA 08 §8.3/8.4.
// ---------------------------------------------------------------------------
__device__ inline void tdm_load_bf16_2d(unsigned lds_addr, const __bf16* gptr,
                                        unsigned tensor_d0, unsigned tensor_d1,
                                        unsigned tile_d0, unsigned tile_d1,
                                        unsigned stride0) {
  const unsigned long long ga = (unsigned long long)(uintptr_t)gptr;
  v4u g0;
  g0.x = 1u;                                    // count=1, user descriptor
  g0.y = lds_addr;                              // LDS byte address
  g0.z = (unsigned)(ga & 0xFFFFFFFFu);          // global addr [95:64]
  g0.w = (unsigned)((ga >> 32) & 0x01FFFFFFu)   // global addr [120:96]
         | (2u << 30);                          // type = 2 (image)
  v8i g1;
  g1[0] = (int)(1u << 16);                      // data_size=1 (2 bytes)
  g1[1] = (int)((tensor_d0 & 0xFFFFu) << 16);   // tensor_dim0 lo16
  g1[2] = (int)(((tensor_d0 >> 16) & 0xFFFFu) | ((tensor_d1 & 0xFFFFu) << 16));
  g1[3] = (int)(((tensor_d1 >> 16) & 0xFFFFu) | ((tile_d0 & 0xFFFFu) << 16));
  g1[4] = (int)(tile_d1 & 0xFFFFu);             // tile_dim1; tile_dim2=0
  g1[5] = (int)stride0;                         // tensor_dim0_stride lo32
  g1[6] = 0;                                    // stride hi16 | dim1_stride lo16
  g1[7] = 0;
  const v4i gz = {0, 0, 0, 0};
#if __clang_major__ >= 23
  const v8i gz8 = {0, 0, 0, 0, 0, 0, 0, 0};
  __builtin_amdgcn_tensor_load_to_lds(g0, g1, gz, gz, gz8, 0);
#else
  __builtin_amdgcn_tensor_load_to_lds(g0, g1, gz, gz, 0);
#endif
}

__device__ inline unsigned lds_off(const void* p) {
  return (unsigned)(uintptr_t)p;  // LDS aperture: flat addr[31:0] = LDS offset
}

// ---------------------------------------------------------------------------
// K1: LayerNorm1 (one block per token)
// ---------------------------------------------------------------------------
__global__ __launch_bounds__(256) void ln1_kernel(const float* __restrict__ x,
                                                  const float* __restrict__ w,
                                                  const float* __restrict__ b,
                                                  float* __restrict__ out) {
  const int tok = blockIdx.x;
  const float* xr = x + (size_t)tok * CB;
  const int tid = threadIdx.x, lane = tid & 31, wv = tid >> 5;
  float v[4], s = 0.f, s2 = 0.f;
#pragma unroll
  for (int i = 0; i < 4; ++i) {
    v[i] = xr[tid + i * 256];
    s += v[i]; s2 += v[i] * v[i];
  }
  __shared__ float red[2][8];
  s = warpSum(s); s2 = warpSum(s2);
  if (lane == 0) { red[0][wv] = s; red[1][wv] = s2; }
  __syncthreads();
  float ts = 0.f, ts2 = 0.f;
#pragma unroll
  for (int k = 0; k < 8; ++k) { ts += red[0][k]; ts2 += red[1][k]; }
  const float mu = ts * (1.0f / CB);
  const float var = ts2 * (1.0f / CB) - mu * mu;
  const float rs = rsqrtf(var + 1e-5f);
#pragma unroll
  for (int i = 0; i < 4; ++i) {
    const int c = tid + i * 256;
    out[(size_t)tok * CB + c] = (v[i] - mu) * rs * w[c] + b[c];
  }
}

// ---------------------------------------------------------------------------
// K2: causal depthwise conv (k=3, pad-left 2) -> bf16 A matrix for QKV GEMM
// ---------------------------------------------------------------------------
__global__ __launch_bounds__(256) void conv_kernel(const float* __restrict__ h,
                                                   const float* __restrict__ cw,
                                                   __bf16* __restrict__ out) {
  const int tok = blockIdx.x;
  const int tt = tok % TT;
#pragma unroll
  for (int i = 0; i < 4; ++i) {
    const int c = threadIdx.x + i * 256;
    const float a0 = (tt >= 2) ? h[(size_t)(tok - 2) * CB + c] : 0.f;
    const float a1 = (tt >= 1) ? h[(size_t)(tok - 1) * CB + c] : 0.f;
    const float a2 = h[(size_t)tok * CB + c];
    const float r = a0 * cw[c * 3 + 0] + a1 * cw[c * 3 + 1] + a2 * cw[c * 3 + 2];
    out[(size_t)tok * CB + c] = (__bf16)r;
  }
}

// ---------------------------------------------------------------------------
// K3: generic GEMM  C[M,N] = A_bf16[M,K] @ B_f32[K,N] (+ residual)
// block 128x128, 8 waves (4x2), wave tile 32x64, K-step 32.
// A tiles: TDM (double buffered, overlapped with WMMA). B tiles: vector loads
// with f32->bf16 convert, double buffered.
// ---------------------------------------------------------------------------
__global__ __launch_bounds__(256, 2) void gemm_bf16(
    const __bf16* __restrict__ A, const float* __restrict__ Bf,
    float* __restrict__ Cout, const float* __restrict__ residual,
    int M, int N, int K) {
  __shared__ __align__(16) __bf16 As[2][128][32];
  __shared__ __align__(16) __bf16 Bst[2][128][32];  // transposed: [n][k]
  const int tid = threadIdx.x, lane = tid & 31, wave = tid >> 5;
  const int lm = lane & 15, hh = lane >> 4;
  const int m0 = blockIdx.y * 128, n0 = blockIdx.x * 128;
  const int wmv = wave & 3, wnv = wave >> 2;
  const int bkp = tid >> 4, bcg = (tid & 15) * 8;   // B copy coords

  auto stageB = [&](int buf, int kk) {
    const int k0 = bkp * 2;
    const float* r0 = Bf + (size_t)(kk + k0) * N + n0 + bcg;
    const float* r1 = r0 + N;
#pragma unroll
    for (int j = 0; j < 8; ++j) {
      Bst[buf][bcg + j][k0]     = (__bf16)r0[j];
      Bst[buf][bcg + j][k0 + 1] = (__bf16)r1[j];
    }
  };

  v8f acc[2][4];
#pragma unroll
  for (int i = 0; i < 2; ++i)
#pragma unroll
    for (int j = 0; j < 4; ++j) acc[i][j] = zero8();

  const int nk = K / 32;
  if (wave == 0)
    tdm_load_bf16_2d(lds_off(&As[0][0][0]), A + (size_t)m0 * K,
                     (unsigned)K, 128u, 32u, 128u, (unsigned)K);
  stageB(0, 0);

  for (int it = 0; it < nk; ++it) {
    const int cur = it & 1, nxt = cur ^ 1;
    const bool hasNext = (it + 1) < nk;
    if (hasNext) {
      const int kk = (it + 1) * 32;
      if (wave == 0)
        tdm_load_bf16_2d(lds_off(&As[nxt][0][0]), A + (size_t)m0 * K + kk,
                         (unsigned)K, 128u, 32u, 128u, (unsigned)K);
      stageB(nxt, kk);
      if (kk + 32 < K)
        __builtin_prefetch(Bf + (size_t)(kk + 32) * N + n0 + bcg, 0, 1);
    }
    if (wave == 0) {
      if (hasNext) __builtin_amdgcn_s_wait_tensorcnt(1);  // current tile done
      else         __builtin_amdgcn_s_wait_tensorcnt(0);
    }
    __syncthreads();  // As[cur] (TDM) + Bst[cur] visible to all waves

    v16bf af[2], bfr[4];
#pragma unroll
    for (int i = 0; i < 2; ++i) {
      const int m = wmv * 32 + i * 16 + lm;
      af[i] = ld2x16(&As[cur][m][hh * 16], &As[cur][m][hh * 16 + 8]);
    }
#pragma unroll
    for (int j = 0; j < 4; ++j) {
      const int n = wnv * 64 + j * 16 + lm;
      bfr[j] = ld2x16(&Bst[cur][n][hh * 16], &Bst[cur][n][hh * 16 + 8]);
    }
#pragma unroll
    for (int i = 0; i < 2; ++i)
#pragma unroll
      for (int j = 0; j < 4; ++j) acc[i][j] = wmma_bf16(af[i], bfr[j], acc[i][j]);
    __syncthreads();  // done reading cur before it is overwritten
  }

#pragma unroll
  for (int i = 0; i < 2; ++i)
#pragma unroll
    for (int j = 0; j < 4; ++j)
#pragma unroll
      for (int e8 = 0; e8 < 8; ++e8) {
        const int row = m0 + wmv * 32 + i * 16 + e8 + 8 * hh;
        const int col = n0 + wnv * 64 + j * 16 + lm;
        float v = acc[i][j][e8];
        if (residual) v += residual[(size_t)row * N + col];
        Cout[(size_t)row * N + col] = v;
      }
}

// ---------------------------------------------------------------------------
// K4: RoPE + repack q/k/v into (B,H,T,D) bf16, q pre-scaled by 1/sqrt(D)
// ---------------------------------------------------------------------------
__global__ __launch_bounds__(256) void rope_kernel(const float* __restrict__ qkv,
                                                   __bf16* __restrict__ Qb,
                                                   __bf16* __restrict__ Kb,
                                                   __bf16* __restrict__ Vb) {
  const int tok = blockIdx.x;
  const int b = tok / TT, t = tok % TT;
  const float* row = qkv + (size_t)tok * (3 * CB);
#pragma unroll
  for (int i = 0; i < 4; ++i) {
    const int c = threadIdx.x + i * 256;
    const int hd = c / DD, d = c % DD, j = d & 31;
    const float inv = __expf(-(float)j * (9.210340371976184f / 32.0f));
    float sn, cs;
    __sincosf((float)t * inv, &sn, &cs);
    const float q = row[c], k = row[CB + c], v = row[2 * CB + c];
    const float qp = (d < 32) ? row[c + 32] : row[c - 32];
    const float kp = (d < 32) ? row[CB + c + 32] : row[CB + c - 32];
    const float sg = (d < 32) ? -1.f : 1.f;
    const size_t o = ((size_t)(b * HH + hd) * TT + t) * DD + d;
    Qb[o] = (__bf16)((q * cs + sg * qp * sn) * 0.125f);
    Kb[o] = (__bf16)(k * cs + sg * kp * sn);
    Vb[o] = (__bf16)v;
  }
}

// ---------------------------------------------------------------------------
// K5: causal flash attention (WMMA). One block per (128 q-rows, b*h).
// 8 waves x 16 q-rows, 32-key steps, online softmax. K tile via TDM.
// ---------------------------------------------------------------------------
__global__ __launch_bounds__(256, 2) void attn_kernel(const __bf16* __restrict__ Qb,
                                                      const __bf16* __restrict__ Kb,
                                                      const __bf16* __restrict__ Vb,
                                                      __bf16* __restrict__ yb) {
  __shared__ __align__(16) __bf16 Qs[128][64];
  __shared__ __align__(16) __bf16 Kt[32][64];   // [key][d]
  __shared__ __align__(16) __bf16 Vt[64][32];   // [d][key]
  __shared__ __align__(16) __bf16 Ps[8][16][32];

  const int tid = threadIdx.x, lane = tid & 31, wave = tid >> 5;
  const int lm = lane & 15, hh = lane >> 4;
  const int bh = blockIdx.y;            // b*H + head
  const int b = bh / HH, head = bh % HH;
  const int q0b = blockIdx.x * 128;
  const size_t hb = (size_t)bh * TT;    // row base in (B,H,T,D)

  // stage Q tile (128x64)
  {
    const int qr = tid >> 1, half = tid & 1;
    const uint4* s = (const uint4*)(Qb + (hb + q0b + qr) * DD + half * 32);
    uint4* d = (uint4*)(&Qs[qr][half * 32]);
    d[0] = s[0]; d[1] = s[1];
  }
  __syncthreads();

  const int q0w = q0b + wave * 16;
  v16bf qf[2];
  {
    const int qr = wave * 16 + lm;
    qf[0] = ld2x16(&Qs[qr][hh * 16],      &Qs[qr][hh * 16 + 8]);
    qf[1] = ld2x16(&Qs[qr][32 + hh * 16], &Qs[qr][32 + hh * 16 + 8]);
  }

  float mrow[8], lrow[8];
#pragma unroll
  for (int e8 = 0; e8 < 8; ++e8) { mrow[e8] = -1e30f; lrow[e8] = 0.f; }
  v8f acc[4];
#pragma unroll
  for (int j = 0; j < 4; ++j) acc[j] = zero8();

  const int nk = (q0b + 128) / 32;      // 32-key steps needed for this block
  for (int kb = 0; kb < nk; ++kb) {
    const int kbase = kb * 32;
    __syncthreads();
    // stage K tile [32][64] = 2048 contiguous bf16 via TDM
    if (wave == 0)
      tdm_load_bf16_2d(lds_off(&Kt[0][0]), Kb + (hb + kbase) * DD,
                       2048u, 1u, 2048u, 1u, 2048u);
    // stage V tile transposed [64][32]
    {
      const int key = tid & 31, dg = (tid >> 5) * 8;
      __bf16 lv[8];
      *(uint4*)lv = *(const uint4*)(Vb + (hb + kbase + key) * DD + dg);
#pragma unroll
      for (int j = 0; j < 8; ++j) Vt[dg + j][key] = lv[j];
    }
    if (wave == 0) __builtin_amdgcn_s_wait_tensorcnt(0);
    __syncthreads();

    if (kbase <= q0w + 15) {  // wave not fully above the diagonal
      v8f s0 = zero8(), s1 = zero8();
#pragma unroll
      for (int c = 0; c < 2; ++c) {
        const int n0k = 0 * 16 + lm, n1k = 1 * 16 + lm;
        v16bf k0 = ld2x16(&Kt[n0k][c * 32 + hh * 16], &Kt[n0k][c * 32 + hh * 16 + 8]);
        v16bf k1 = ld2x16(&Kt[n1k][c * 32 + hh * 16], &Kt[n1k][c * 32 + hh * 16 + 8]);
        s0 = wmma_bf16(qf[c], k0, s0);
        s1 = wmma_bf16(qf[c], k1, s1);
      }
      const int cg0 = kbase + lm, cg1 = kbase + 16 + lm;
      float corr[8];
#pragma unroll
      for (int e8 = 0; e8 < 8; ++e8) {
        const int rowg = q0w + e8 + 8 * hh;
        float a = s0[e8], bsc = s1[e8];
        if (cg0 > rowg) a = -1e30f;
        if (cg1 > rowg) bsc = -1e30f;
        float mx = fmaxf(a, bsc);
#pragma unroll
        for (int msk = 1; msk <= 8; msk <<= 1) mx = fmaxf(mx, __shfl_xor(mx, msk));
        const float mn = fmaxf(mrow[e8], mx);
        const float p0 = __expf(a - mn), p1 = __expf(bsc - mn);
        float rs = p0 + p1;
#pragma unroll
        for (int msk = 1; msk <= 8; msk <<= 1) rs += __shfl_xor(rs, msk);
        corr[e8] = __expf(mrow[e8] - mn);
        lrow[e8] = lrow[e8] * corr[e8] + rs;
        mrow[e8] = mn;
        Ps[wave][e8 + 8 * hh][lm]      = (__bf16)p0;
        Ps[wave][e8 + 8 * hh][16 + lm] = (__bf16)p1;
      }
#pragma unroll
      for (int j = 0; j < 4; ++j)
#pragma unroll
        for (int e8 = 0; e8 < 8; ++e8) acc[j][e8] *= corr[e8];

      // same-wave LDS RAW: P store -> P fragment load
      asm volatile("s_wait_dscnt 0" ::: "memory");
      v16bf pf = ld2x16(&Ps[wave][lm][hh * 16], &Ps[wave][lm][hh * 16 + 8]);
#pragma unroll
      for (int j = 0; j < 4; ++j) {
        const int dr = j * 16 + lm;
        v16bf vf = ld2x16(&Vt[dr][hh * 16], &Vt[dr][hh * 16 + 8]);
        acc[j] = wmma_bf16(pf, vf, acc[j]);
      }
    }
  }

  // write y as bf16 (B*T, C) with col = head*64 + d
#pragma unroll
  for (int j = 0; j < 4; ++j)
#pragma unroll
    for (int e8 = 0; e8 < 8; ++e8) {
      const int q = q0w + e8 + 8 * hh;
      const int col = head * DD + j * 16 + lm;
      yb[(size_t)(b * TT + q) * CB + col] = (__bf16)(acc[j][e8] / lrow[e8]);
    }
}

// ---------------------------------------------------------------------------
// K6: zero routing metadata + init permutation padding
// ---------------------------------------------------------------------------
__global__ void moe_init(int* __restrict__ perm, int* __restrict__ counts,
                         int* __restrict__ cursor) {
  const int i = blockIdx.x * blockDim.x + threadIdx.x;
  if (i < PADM) perm[i] = -1;
  if (i < EE) { counts[i] = 0; cursor[i] = 0; }
}

// ---------------------------------------------------------------------------
// K7: LayerNorm2 + router argmax + expert counts; writes h2 as bf16
// ---------------------------------------------------------------------------
__global__ __launch_bounds__(256) void ln2_router(const float* __restrict__ x1,
                                                  const float* __restrict__ w,
                                                  const float* __restrict__ b,
                                                  const float* __restrict__ rw,
                                                  __bf16* __restrict__ h2b,
                                                  int* __restrict__ eidx,
                                                  int* __restrict__ counts) {
  const int tok = blockIdx.x;
  const float* xr = x1 + (size_t)tok * CB;
  const int tid = threadIdx.x, lane = tid & 31, wv = tid >> 5;
  float v[4], s = 0.f, s2 = 0.f;
#pragma unroll
  for (int i = 0; i < 4; ++i) {
    v[i] = xr[tid + i * 256];
    s += v[i]; s2 += v[i] * v[i];
  }
  __shared__ float red[2][8];
  s = warpSum(s); s2 = warpSum(s2);
  if (lane == 0) { red[0][wv] = s; red[1][wv] = s2; }
  __syncthreads();
  float ts = 0.f, ts2 = 0.f;
#pragma unroll
  for (int k = 0; k < 8; ++k) { ts += red[0][k]; ts2 += red[1][k]; }
  const float mu = ts * (1.0f / CB);
  const float rs = rsqrtf(ts2 * (1.0f / CB) - mu * mu + 1e-5f);

  float part[EE] = {0.f, 0.f, 0.f, 0.f};
#pragma unroll
  for (int i = 0; i < 4; ++i) {
    const int c = tid + i * 256;
    const float hv = (v[i] - mu) * rs * w[c] + b[c];
    h2b[(size_t)tok * CB + c] = (__bf16)hv;
#pragma unroll
    for (int e = 0; e < EE; ++e) part[e] += hv * rw[c * EE + e];
  }
  __shared__ float rp[EE][8];
#pragma unroll
  for (int e = 0; e < EE; ++e) {
    const float pe = warpSum(part[e]);
    if (lane == 0) rp[e][wv] = pe;
  }
  __syncthreads();
  if (tid == 0) {
    float best = -1e30f; int be = 0;
#pragma unroll
    for (int e = 0; e < EE; ++e) {
      float sl = 0.f;
#pragma unroll
      for (int k = 0; k < 8; ++k) sl += rp[e][k];
      if (sl > best) { best = sl; be = e; }
    }
    eidx[tok] = be;
    atomicAdd(&counts[be], 1);
  }
}

// ---------------------------------------------------------------------------
// K8: plan expert segments (padded to 128) + per-tile expert map
// ---------------------------------------------------------------------------
__global__ void moe_plan(const int* __restrict__ counts, int* __restrict__ seg_start,
                         int* __restrict__ tile_expert) {
  if (threadIdx.x == 0 && blockIdx.x == 0) {
    int starts[EE], ends[EE], off = 0;
    for (int e = 0; e < EE; ++e) {
      starts[e] = off;
      off += ((counts[e] + 127) / 128) * 128;
      ends[e] = off;
      seg_start[e] = starts[e];
    }
    for (int m = 0; m < MT_TILES; ++m) {
      const int r = m * 128;
      int ex = -1;
      for (int e = 0; e < EE; ++e)
        if (r >= starts[e] && r < ends[e]) ex = e;
      tile_expert[m] = ex;
    }
  }
}

// ---------------------------------------------------------------------------
// K9: scatter token ids into their expert segment
// ---------------------------------------------------------------------------
__global__ void moe_scatter(const int* __restrict__ eidx,
                            const int* __restrict__ seg_start,
                            int* __restrict__ cursor, int* __restrict__ perm) {
  const int i = blockIdx.x * blockDim.x + threadIdx.x;
  if (i < MTOK) {
    const int e = eidx[i];
    const int pos = seg_start[e] + atomicAdd(&cursor[e], 1);
    perm[pos] = i;
  }
}

// ---------------------------------------------------------------------------
// K10: MoE GEMM1: u = silu(h2@Wg) * (h2@Wv), gathered rows, fused gate/value.
// Block: 128 rows x 64 u-cols; B tile holds 64 gate + 64 value columns.
// (A rows are permutation-gathered -> vector staging, not TDM.)
// ---------------------------------------------------------------------------
__global__ __launch_bounds__(256, 2) void moe_gemm1(
    const __bf16* __restrict__ h2b, const float* __restrict__ expert_wv,
    const int* __restrict__ perm, const int* __restrict__ tile_expert,
    __bf16* __restrict__ U) {
  const int ex = tile_expert[blockIdx.y];
  if (ex < 0) return;
  const float* Bf = expert_wv + (size_t)ex * CB * (2 * DFF);

  __shared__ __align__(16) __bf16 As[128][32];
  __shared__ __align__(16) __bf16 Bst[128][32];  // [0..63]=gate, [64..127]=value
  const int tid = threadIdx.x, lane = tid & 31, wave = tid >> 5;
  const int lm = lane & 15, hh = lane >> 4;
  const int m0 = blockIdx.y * 128, n0 = blockIdx.x * 64;
  const int wmv = wave & 3, wnv = wave >> 2;
  const int am = tid >> 1, akh = tid & 1;
  const int bkp = tid >> 4, bcg = (tid & 15) * 8;
  const int tokA = perm[m0 + am];

  v8f ag[2][2], av[2][2];
#pragma unroll
  for (int i = 0; i < 2; ++i)
#pragma unroll
    for (int j = 0; j < 2; ++j) { ag[i][j] = zero8(); av[i][j] = zero8(); }

  for (int kk = 0; kk < CB; kk += 32) {
    __syncthreads();
    {
      uint4* d = (uint4*)(&As[am][akh * 16]);
      if (tokA >= 0) {
        const uint4* s = (const uint4*)(h2b + (size_t)tokA * CB + kk + akh * 16);
        d[0] = s[0]; d[1] = s[1];
      } else {
        const uint4 z = {0u, 0u, 0u, 0u};
        d[0] = z; d[1] = z;
      }
    }
    {
      const int k0 = bkp * 2;
#pragma unroll
      for (int j = 0; j < 8; ++j) {
        const int cc = bcg + j;
        const int gcol = (cc < 64) ? (n0 + cc) : (DFF + n0 + (cc - 64));
        Bst[cc][k0]     = (__bf16)Bf[(size_t)(kk + k0) * (2 * DFF) + gcol];
        Bst[cc][k0 + 1] = (__bf16)Bf[(size_t)(kk + k0 + 1) * (2 * DFF) + gcol];
      }
    }
    __syncthreads();

    v16bf af[2];
#pragma unroll
    for (int i = 0; i < 2; ++i) {
      const int m = wmv * 32 + i * 16 + lm;
      af[i] = ld2x16(&As[m][hh * 16], &As[m][hh * 16 + 8]);
    }
#pragma unroll
    for (int j = 0; j < 2; ++j) {
      const int ng = wnv * 32 + j * 16 + lm;
      v16bf bg = ld2x16(&Bst[ng][hh * 16], &Bst[ng][hh * 16 + 8]);
      v16bf bv = ld2x16(&Bst[64 + ng][hh * 16], &Bst[64 + ng][hh * 16 + 8]);
#pragma unroll
      for (int i = 0; i < 2; ++i) {
        ag[i][j] = wmma_bf16(af[i], bg, ag[i][j]);
        av[i][j] = wmma_bf16(af[i], bv, av[i][j]);
      }
    }
  }

#pragma unroll
  for (int i = 0; i < 2; ++i)
#pragma unroll
    for (int j = 0; j < 2; ++j)
#pragma unroll
      for (int e8 = 0; e8 < 8; ++e8) {
        const float g = ag[i][j][e8], vv = av[i][j][e8];
        const float u = g * (1.0f / (1.0f + __expf(-g))) * vv;
        const int row = m0 + wmv * 32 + i * 16 + e8 + 8 * hh;
        const int col = n0 + wnv * 32 + j * 16 + lm;
        U[(size_t)row * DFF + col] = (__bf16)u;
      }
}

// ---------------------------------------------------------------------------
// K11: MoE GEMM2: out[token] = x1[token] + u @ expert_proj[e], scatter by perm
// A tiles (U, bf16, dense rows) via TDM double-buffered.
// ---------------------------------------------------------------------------
__global__ __launch_bounds__(256, 2) void moe_gemm2(
    const __bf16* __restrict__ U, const float* __restrict__ expert_proj,
    const int* __restrict__ perm, const int* __restrict__ tile_expert,
    const float* __restrict__ x1, float* __restrict__ out) {
  const int ex = tile_expert[blockIdx.y];
  if (ex < 0) return;
  const float* Bf = expert_proj + (size_t)ex * DFF * CB;

  __shared__ __align__(16) __bf16 As[2][128][32];
  __shared__ __align__(16) __bf16 Bst[2][128][32];
  const int tid = threadIdx.x, lane = tid & 31, wave = tid >> 5;
  const int lm = lane & 15, hh = lane >> 4;
  const int m0 = blockIdx.y * 128, n0 = blockIdx.x * 128;
  const int wmv = wave & 3, wnv = wave >> 2;
  const int bkp = tid >> 4, bcg = (tid & 15) * 8;

  auto stageB = [&](int buf, int kk) {
    const int k0 = bkp * 2;
    const float* r0 = Bf + (size_t)(kk + k0) * CB + n0 + bcg;
    const float* r1 = r0 + CB;
#pragma unroll
    for (int j = 0; j < 8; ++j) {
      Bst[buf][bcg + j][k0]     = (__bf16)r0[j];
      Bst[buf][bcg + j][k0 + 1] = (__bf16)r1[j];
    }
  };

  v8f acc[2][4];
#pragma unroll
  for (int i = 0; i < 2; ++i)
#pragma unroll
    for (int j = 0; j < 4; ++j) acc[i][j] = zero8();

  const int nk = DFF / 32;
  if (wave == 0)
    tdm_load_bf16_2d(lds_off(&As[0][0][0]), U + (size_t)m0 * DFF,
                     (unsigned)DFF, 128u, 32u, 128u, (unsigned)DFF);
  stageB(0, 0);

  for (int it = 0; it < nk; ++it) {
    const int cur = it & 1, nxt = cur ^ 1;
    const bool hasNext = (it + 1) < nk;
    if (hasNext) {
      const int kk = (it + 1) * 32;
      if (wave == 0)
        tdm_load_bf16_2d(lds_off(&As[nxt][0][0]), U + (size_t)m0 * DFF + kk,
                         (unsigned)DFF, 128u, 32u, 128u, (unsigned)DFF);
      stageB(nxt, kk);
      if (kk + 32 < DFF)
        __builtin_prefetch(Bf + (size_t)(kk + 32) * CB + n0 + bcg, 0, 1);
    }
    if (wave == 0) {
      if (hasNext) __builtin_amdgcn_s_wait_tensorcnt(1);
      else         __builtin_amdgcn_s_wait_tensorcnt(0);
    }
    __syncthreads();

    v16bf af[2], bfr[4];
#pragma unroll
    for (int i = 0; i < 2; ++i) {
      const int m = wmv * 32 + i * 16 + lm;
      af[i] = ld2x16(&As[cur][m][hh * 16], &As[cur][m][hh * 16 + 8]);
    }
#pragma unroll
    for (int j = 0; j < 4; ++j) {
      const int n = wnv * 64 + j * 16 + lm;
      bfr[j] = ld2x16(&Bst[cur][n][hh * 16], &Bst[cur][n][hh * 16 + 8]);
    }
#pragma unroll
    for (int i = 0; i < 2; ++i)
#pragma unroll
      for (int j = 0; j < 4; ++j) acc[i][j] = wmma_bf16(af[i], bfr[j], acc[i][j]);
    __syncthreads();
  }

  int tokr[2][8];
#pragma unroll
  for (int i = 0; i < 2; ++i)
#pragma unroll
    for (int e8 = 0; e8 < 8; ++e8)
      tokr[i][e8] = perm[m0 + wmv * 32 + i * 16 + e8 + 8 * hh];

#pragma unroll
  for (int i = 0; i < 2; ++i)
#pragma unroll
    for (int j = 0; j < 4; ++j)
#pragma unroll
      for (int e8 = 0; e8 < 8; ++e8) {
        const int tok = tokr[i][e8];
        if (tok >= 0) {
          const int col = n0 + wnv * 64 + j * 16 + lm;
          out[(size_t)tok * CB + col] =
              x1[(size_t)tok * CB + col] + acc[i][j][e8];
        }
      }
}

// ---------------------------------------------------------------------------
// Host-side orchestration
// ---------------------------------------------------------------------------
extern "C" void kernel_launch(void* const* d_in, const int* in_sizes, int n_in,
                              void* d_out, int out_size, void* d_ws, size_t ws_size,
                              hipStream_t stream) {
  const float* x          = (const float*)d_in[0];
  const float* ln1_w      = (const float*)d_in[1];
  const float* ln1_b      = (const float*)d_in[2];
  const float* ln2_w      = (const float*)d_in[3];
  const float* ln2_b      = (const float*)d_in[4];
  const float* conv_w     = (const float*)d_in[5];
  const float* attn_w     = (const float*)d_in[6];
  const float* proj_w     = (const float*)d_in[7];
  const float* router_w   = (const float*)d_in[8];
  const float* expert_wv  = (const float*)d_in[9];
  const float* expert_proj= (const float*)d_in[10];
  float* out = (float*)d_out;

  char* w = (char*)d_ws;
  auto alloc = [&](size_t bytes) -> void* {
    void* p = (void*)w;
    w += (bytes + 255) & ~(size_t)255;
    return p;
  };
  float*  h_ln = (float*)alloc((size_t)MTOK * CB * 4);
  __bf16* hbf  = (__bf16*)alloc((size_t)MTOK * CB * 2);
  float*  qkv  = (float*)alloc((size_t)MTOK * 3 * CB * 4);
  __bf16* Qb   = (__bf16*)alloc((size_t)MTOK * CB * 2);
  __bf16* Kb   = (__bf16*)alloc((size_t)MTOK * CB * 2);
  __bf16* Vb   = (__bf16*)alloc((size_t)MTOK * CB * 2);
  __bf16* yb   = (__bf16*)alloc((size_t)MTOK * CB * 2);
  float*  x1   = (float*)alloc((size_t)MTOK * CB * 4);
  __bf16* h2b  = (__bf16*)alloc((size_t)MTOK * CB * 2);
  __bf16* U    = (__bf16*)alloc((size_t)PADM * DFF * 2);
  int* counts      = (int*)alloc(EE * 4);
  int* cursor      = (int*)alloc(EE * 4);
  int* seg_start   = (int*)alloc(EE * 4);
  int* tile_expert = (int*)alloc(MT_TILES * 4);
  int* eidx        = (int*)alloc((size_t)MTOK * 4);
  int* perm        = (int*)alloc((size_t)PADM * 4);

  // attention path
  ln1_kernel<<<MTOK, 256, 0, stream>>>(x, ln1_w, ln1_b, h_ln);
  conv_kernel<<<MTOK, 256, 0, stream>>>(h_ln, conv_w, hbf);
  gemm_bf16<<<dim3(3 * CB / 128, MTOK / 128), 256, 0, stream>>>(
      hbf, attn_w, qkv, nullptr, MTOK, 3 * CB, CB);
  rope_kernel<<<MTOK, 256, 0, stream>>>(qkv, Qb, Kb, Vb);
  attn_kernel<<<dim3(TT / 128, BB * HH), 256, 0, stream>>>(Qb, Kb, Vb, yb);
  gemm_bf16<<<dim3(CB / 128, MTOK / 128), 256, 0, stream>>>(
      yb, proj_w, x1, x, MTOK, CB, CB);

  // MoE path (top-1 routed, expert-gathered)
  moe_init<<<(PADM + 255) / 256, 256, 0, stream>>>(perm, counts, cursor);
  ln2_router<<<MTOK, 256, 0, stream>>>(x1, ln2_w, ln2_b, router_w, h2b, eidx, counts);
  moe_plan<<<1, 32, 0, stream>>>(counts, seg_start, tile_expert);
  moe_scatter<<<MTOK / 256, 256, 0, stream>>>(eidx, seg_start, cursor, perm);
  moe_gemm1<<<dim3(DFF / 64, MT_TILES), 256, 0, stream>>>(h2b, expert_wv, perm,
                                                          tile_expert, U);
  moe_gemm2<<<dim3(CB / 128, MT_TILES), 256, 0, stream>>>(U, expert_proj, perm,
                                                          tile_expert, x1, out);
}
